// AudioModel_26800595927710
// MI455X (gfx1250) — compile-verified, matrix operand order checked by reading, not run on hardware
//
#include <hip/hip_runtime.h>

typedef __attribute__((ext_vector_type(16))) _Float16 v16h;
typedef __attribute__((ext_vector_type(8)))  float    v8f;
typedef __attribute__((ext_vector_type(2)))  _Float16 v2h;

#define B_   64
#define T_   2048
#define NIN  257
#define H_   32
#define G4   128   // 4*HIDDEN
#define NOUT 10

// K index of the first element of the packed f16 pair held in operand VGPR `v`
// for lane-half `hi` (0 = lanes 0-15, 1 = lanes 16-31), per the CDNA5 ISA
// 16-bit A/B 16x32 layout (CDNA5 ISA §7.12.2).
__device__ __forceinline__ int kmap(int v, int hi) {
    int base8 = hi * 8;
    return (v < 4) ? (base8 + 2 * v) : (16 + base8 + 2 * (v - 4));
}

__device__ __forceinline__ float sigm(float x) {
    return 1.0f / (1.0f + __expf(-x));
}
__device__ __forceinline__ float tanh_fast(float x) {
    float e = __expf(2.0f * x);
    return (e - 1.0f) / (e + 1.0f);
}

// ---------------------------------------------------------------------------
// Kernel 0: one-shot f32 -> f16 weight conversion into scratch.
//   wih16: [128][256] (only k<256 consumed by WMMA; k=256 handled in f32)
//   whh16: [128][32]
// ---------------------------------------------------------------------------
__global__ void convert_weights(const float* __restrict__ Wih,
                                const float* __restrict__ Whh,
                                _Float16* __restrict__ wih16,
                                _Float16* __restrict__ whh16) {
    int idx = blockIdx.x * 256 + threadIdx.x;
    if (idx < G4 * 256) {
        int n = idx >> 8, k = idx & 255;
        wih16[idx] = (_Float16)Wih[n * NIN + k];
    } else if (idx < G4 * 256 + G4 * H_) {
        int i = idx - G4 * 256;
        whh16[i] = (_Float16)Whh[i];
    }
}

// ---------------------------------------------------------------------------
// Kernel 1: x_proj[t][b][n] = sum_k specs[b][t][k] * W_ih[n][k] + b_ih[n]+b_hh[n]
// One wave = one 16-row M tile x all 128 columns (8 WMMA accumulators).
// K loop: 8 chunks of 32 via v_wmma_f32_16x16x32_f16; k==256 column + bias
// folded into the f32 accumulator init. Output stored time-major [T,B,128].
// ---------------------------------------------------------------------------
__global__ __launch_bounds__(256)
void xproj_gemm(const float* __restrict__ specs,
                const float* __restrict__ Wih,
                const _Float16* __restrict__ wih16,
                const float* __restrict__ b_ih,
                const float* __restrict__ b_hh,
                float* __restrict__ xproj) {
    const int tid  = threadIdx.x;
    const int wave = tid >> 5, lane = tid & 31;
    const int lm = lane & 15, hi = lane >> 4;
    const int mbase = (blockIdx.x * 8 + wave) * 16;

    // remainder column k=256 (f32 exact) + bias, folded into C init
    float s256[8];
    #pragma unroll
    for (int r = 0; r < 8; ++r)
        s256[r] = specs[(size_t)(mbase + hi * 8 + r) * NIN + 256];

    v8f c[8];
    #pragma unroll
    for (int nt = 0; nt < 8; ++nt) {
        int n = nt * 16 + lm;
        float bias = b_ih[n] + b_hh[n];
        float w256 = Wih[n * NIN + 256];
        #pragma unroll
        for (int r = 0; r < 8; ++r) c[nt][r] = bias + s256[r] * w256;
    }

    const float* arow = specs + (size_t)(mbase + lm) * NIN;
    #pragma unroll 1
    for (int kk = 0; kk < 8; ++kk) {
        const int kb = kk * 32;
        __builtin_prefetch(arow + kb + 32, 0, 1);  // next A chunk -> global_prefetch
        v16h a;
        #pragma unroll
        for (int v = 0; v < 8; ++v) {
            int k0 = kmap(v, hi);
            a[2 * v]     = (_Float16)arow[kb + k0];
            a[2 * v + 1] = (_Float16)arow[kb + k0 + 1];
        }
        #pragma unroll
        for (int nt = 0; nt < 8; ++nt) {
            int n = nt * 16 + lm;
            const _Float16* brow = wih16 + n * 256 + kb;
            v16h b;
            #pragma unroll
            for (int v = 0; v < 8; ++v) {
                int k0 = kmap(v, hi);
                v2h p = *(const v2h*)(brow + k0);
                b[2 * v]     = p.x;
                b[2 * v + 1] = p.y;
            }
            c[nt] = __builtin_amdgcn_wmma_f32_16x16x32_f16(
                false, a, false, b, (short)0, c[nt], false, false);
        }
    }

    // store time-major: xproj[(t*B + b)*128 + n]
    #pragma unroll
    for (int nt = 0; nt < 8; ++nt) {
        int n = nt * 16 + lm;
        #pragma unroll
        for (int r = 0; r < 8; ++r) {
            int m  = mbase + hi * 8 + r;
            int bi = m >> 11;          // m / T
            int tt = m & (T_ - 1);     // m % T
            xproj[((size_t)tt * B_ + bi) * G4 + n] = c[nt][r];
        }
    }
}

// ---------------------------------------------------------------------------
// Kernel 2: sequential LSTM scan. One workgroup (8 waves) per 16 batch rows;
// each wave owns one 16-column N tile of W_hh^T in registers for all 2048
// steps and issues one v_wmma per step (M=16 batch, K=32 hidden). Cell state
// lives in registers; h is exchanged through LDS (f16 for the A operand).
// Finishes with the ReLU head + log-softmax.
// ---------------------------------------------------------------------------
__global__ __launch_bounds__(256)
void lstm_scan(const float* __restrict__ xproj,
               const _Float16* __restrict__ whh16,
               const float* __restrict__ W_out,
               const float* __restrict__ b_out,
               float* __restrict__ out) {
    __shared__ _Float16 h16[16 * H_];   // hidden state, f16 (WMMA A source)
    __shared__ float    gates[16 * G4]; // 16 x 128 pre-activations
    __shared__ float    hf32[16 * H_];  // f32 hidden for the output head

    const int tid  = threadIdx.x;
    const int nt   = tid >> 5, lane = tid & 31;
    const int lm   = lane & 15, hi = lane >> 4;
    const int bb   = blockIdx.x * 16;   // global batch base of this workgroup

    // Preload constant B tile = W_hh^T columns [nt*16, nt*16+16)
    v16h bt;
    {
        const _Float16* brow = whh16 + (nt * 16 + lm) * H_;
        #pragma unroll
        for (int v = 0; v < 8; ++v) {
            int k0 = kmap(v, hi);
            v2h p = *(const v2h*)(brow + k0);
            bt[2 * v]     = p.x;
            bt[2 * v + 1] = p.y;
        }
    }

    // h = 0, c = 0. Each thread owns cells (bA, lane) and (bB, lane).
    h16[tid]       = (_Float16)0.0f;
    h16[tid + 256] = (_Float16)0.0f;
    const int bA = nt, bB = nt + 8, jb = lane;
    float cA = 0.0f, cB = 0.0f;
    __syncthreads();

    #pragma unroll 1
    for (int t = 0; t < T_; ++t) {
        // A tile: current h (16x32 f16) from LDS in WMMA lane layout
        v16h a;
        const _Float16* hrow = h16 + lm * H_;
        #pragma unroll
        for (int v = 0; v < 8; ++v) {
            int k0 = kmap(v, hi);
            a[2 * v]     = hrow[k0];
            a[2 * v + 1] = hrow[k0 + 1];
        }
        // C init: precomputed input projection (bias already folded in)
        v8f c;
        const float* xrow = xproj + ((size_t)t * B_ + bb) * G4 + nt * 16 + lm;
        #pragma unroll
        for (int r = 0; r < 8; ++r) c[r] = xrow[(size_t)(hi * 8 + r) * G4];

        c = __builtin_amdgcn_wmma_f32_16x16x32_f16(
            false, a, false, bt, (short)0, c, false, false);

        #pragma unroll
        for (int r = 0; r < 8; ++r)
            gates[(hi * 8 + r) * G4 + nt * 16 + lm] = c[r];
        __syncthreads();

        // Elementwise LSTM cell update: 2 (b,j) cells per thread
        {
            const float* gA = gates + bA * G4 + jb;
            float i0 = sigm(gA[0]);
            float f0 = sigm(gA[32]);
            float g0 = tanh_fast(gA[64]);
            float o0 = sigm(gA[96]);
            cA = f0 * cA + i0 * g0;
            float hA = o0 * tanh_fast(cA);
            h16[bA * H_ + jb]  = (_Float16)hA;
            hf32[bA * H_ + jb] = hA;

            const float* gB = gates + bB * G4 + jb;
            float i1 = sigm(gB[0]);
            float f1 = sigm(gB[32]);
            float g1 = tanh_fast(gB[64]);
            float o1 = sigm(gB[96]);
            cB = f1 * cB + i1 * g1;
            float hB = o1 * tanh_fast(cB);
            h16[bB * H_ + jb]  = (_Float16)hB;
            hf32[bB * H_ + jb] = hB;
        }
        __syncthreads();
    }

    // Output head: logits = relu(h) @ W_out^T + b_out, then log_softmax
    if (tid < 16) {
        float hrelu[H_];
        #pragma unroll
        for (int j = 0; j < H_; ++j) {
            float v = hf32[tid * H_ + j];
            hrelu[j] = v > 0.0f ? v : 0.0f;
        }
        float logit[NOUT];
        float mx = -1e30f;
        #pragma unroll
        for (int o = 0; o < NOUT; ++o) {
            float acc = b_out[o];
            #pragma unroll
            for (int j = 0; j < H_; ++j) acc += hrelu[j] * W_out[o * H_ + j];
            logit[o] = acc;
            mx = fmaxf(mx, acc);
        }
        float se = 0.0f;
        #pragma unroll
        for (int o = 0; o < NOUT; ++o) se += __expf(logit[o] - mx);
        float lse = __logf(se) + mx;
        #pragma unroll
        for (int o = 0; o < NOUT; ++o)
            out[(bb + tid) * NOUT + o] = logit[o] - lse;
    }
}

extern "C" void kernel_launch(void* const* d_in, const int* in_sizes, int n_in,
                              void* d_out, int out_size, void* d_ws, size_t ws_size,
                              hipStream_t stream) {
    const float* specs = (const float*)d_in[0];
    const float* W_ih  = (const float*)d_in[1];
    const float* W_hh  = (const float*)d_in[2];
    const float* b_ih  = (const float*)d_in[3];
    const float* b_hh  = (const float*)d_in[4];
    const float* W_out = (const float*)d_in[5];
    const float* b_out = (const float*)d_in[6];
    float* out = (float*)d_out;

    // Scratch layout: x_proj (time-major, 64 MiB) + f16 weights (72 KiB)
    char* ws = (char*)d_ws;
    float*    xproj = (float*)ws;
    _Float16* wih16 = (_Float16*)(ws + (size_t)B_ * T_ * G4 * sizeof(float));
    _Float16* whh16 = (_Float16*)((char*)wih16 + (size_t)G4 * 256 * sizeof(_Float16));

    convert_weights<<<(G4 * 256 + G4 * H_ + 255) / 256, 256, 0, stream>>>(
        W_ih, W_hh, wih16, whh16);
    // (B*T)/16 M-tiles, 8 tiles (8 waves) per block -> 1024 blocks
    xproj_gemm<<<(B_ * T_) / (16 * 8), 256, 0, stream>>>(
        specs, W_ih, wih16, b_ih, b_hh, xproj);
    // 4 workgroups of 16 batch rows each run the sequential scan in parallel
    lstm_scan<<<B_ / 16, 256, 0, stream>>>(xproj, whh16, W_out, b_out, out);
}
